// TinyLoRALinear_3092376453745
// MI455X (gfx1250) — compile-verified, matrix operand order checked by reading, not run on hardware
//
#include <hip/hip_runtime.h>

// ---------------------------------------------------------------------------
// TinyLoRA linear, folded:  out = x @ (W + (U@S@R) @ Vh)^T + b
//   x: [8192, 4096] f32   W: [4096, 4096] f32   out: [8192, 4096] f32
// Fold rank-2 update into W, split x and W' into bf16 hi/lo pairs (one-time
// conversion into workspace), then one WMMA bf16 GEMM with 3-product split
// accumulation (hi*hi + lo*hi + hi*lo) in f32. Register double-buffered.
// ---------------------------------------------------------------------------

#define M_DIM 8192
#define N_DIM 4096
#define K_DIM 4096
#define RANK  2
#define UDIM  16

typedef __bf16 bf16_t;
typedef __attribute__((ext_vector_type(4)))  __bf16 v4bf;
typedef __attribute__((ext_vector_type(8)))  __bf16 v8bf;
typedef __attribute__((ext_vector_type(16))) __bf16 v16bf;
typedef __attribute__((ext_vector_type(8)))  float  v8f;
typedef __attribute__((ext_vector_type(4)))  float  v4f;

// ---------------------------------------------------------------------------
// Kernel 1: M2[o,r] = sum_c U[o,c] * (S @ R)[c,r],  R = sum_u v[u] * P[u]
// ---------------------------------------------------------------------------
__global__ __launch_bounds__(256) void tinylora_prep_m2(
    const float* __restrict__ U, const float* __restrict__ S,
    const float* __restrict__ P, const float* __restrict__ v,
    float* __restrict__ M2) {
  int o = blockIdx.x * 256 + threadIdx.x;
  float r00 = 0.f, r01 = 0.f, r10 = 0.f, r11 = 0.f;
#pragma unroll
  for (int u = 0; u < UDIM; ++u) {
    float vu = v[u];
    const float* p = P + u * RANK * RANK;
    r00 += vu * p[0]; r01 += vu * p[1];
    r10 += vu * p[2]; r11 += vu * p[3];
  }
  float s00 = S[0], s01 = S[1], s10 = S[2], s11 = S[3];
  float sr00 = s00 * r00 + s01 * r10;
  float sr01 = s00 * r01 + s01 * r11;
  float sr10 = s10 * r00 + s11 * r10;
  float sr11 = s10 * r01 + s11 * r11;
  float u0 = U[o * RANK + 0], u1 = U[o * RANK + 1];
  M2[o * RANK + 0] = u0 * sr00 + u1 * sr10;
  M2[o * RANK + 1] = u0 * sr01 + u1 * sr11;
}

// ---------------------------------------------------------------------------
// Kernel 2: W'[o,d] = W[o,d] + M2[o,0]*Vh[0,d] + M2[o,1]*Vh[1,d] -> bf16 hi/lo
// W is read once: non-temporal. w_hi/w_lo are hot (GEMM reuse): regular store.
// ---------------------------------------------------------------------------
__global__ __launch_bounds__(256) void tinylora_fold_w(
    const float* __restrict__ W, const float* __restrict__ Vh,
    const float* __restrict__ M2,
    bf16_t* __restrict__ w_hi, bf16_t* __restrict__ w_lo) {
  int t = blockIdx.x * 256 + threadIdx.x;     // 4,194,304 threads
  int o = t >> 10;                            // K/4 = 1024 quads per row
  int d = (t & 1023) << 2;
  size_t base = (size_t)o * K_DIM + d;
  v4f w  = __builtin_nontemporal_load((const v4f*)(W + base));
  v4f v0 = *(const v4f*)(Vh + d);
  v4f v1 = *(const v4f*)(Vh + K_DIM + d);
  float m0 = M2[o * RANK + 0], m1 = M2[o * RANK + 1];
  v4bf hi, lo;
#pragma unroll
  for (int i = 0; i < 4; ++i) {
    float f = w[i] + m0 * v0[i] + m1 * v1[i];
    __bf16 h = (__bf16)f;
    hi[i] = h;
    lo[i] = (__bf16)(f - (float)h);
  }
  *(v4bf*)(w_hi + base) = hi;
  *(v4bf*)(w_lo + base) = lo;
}

// ---------------------------------------------------------------------------
// Kernel 3: split x into bf16 hi/lo. x read once: non-temporal.
// ---------------------------------------------------------------------------
__global__ __launch_bounds__(256) void tinylora_cvt_x(
    const float* __restrict__ x,
    bf16_t* __restrict__ x_hi, bf16_t* __restrict__ x_lo) {
  size_t t = (size_t)blockIdx.x * 256 + threadIdx.x;  // 8,388,608 threads
  size_t base = t * 4;
  v4f w = __builtin_nontemporal_load((const v4f*)(x + base));
  v4bf hi, lo;
#pragma unroll
  for (int i = 0; i < 4; ++i) {
    float f = w[i];
    __bf16 h = (__bf16)f;
    hi[i] = h;
    lo[i] = (__bf16)(f - (float)h);
  }
  *(v4bf*)(x_hi + base) = hi;
  *(v4bf*)(x_lo + base) = lo;
}

// ---------------------------------------------------------------------------
// Fragment loader: uniform SGPR base + 32-bit lane byte offset (GVS mode),
// chunk deltas fold into the 24-bit instruction immediate.
// Per the ISA 16-bit A 16x32 layout (B 32x16 is its symmetric transpose):
// lane holds two contiguous 8-element K-chunks at
// k + (lane<16?0:8) and k + 16 + (lane<16?0:8). NT GEMM => both operands
// K-contiguous => two global_load_b128 per fragment.
// ---------------------------------------------------------------------------
static __device__ __forceinline__ v16bf ldfrag(const char* __restrict__ base,
                                               unsigned off) {
  v8bf c0 = *(const v8bf*)(base + off);
  v8bf c1 = *(const v8bf*)(base + off + 32);
  return __builtin_shufflevector(c0, c1, 0, 1, 2, 3, 4, 5, 6, 7,
                                 8, 9, 10, 11, 12, 13, 14, 15);
}

static __device__ __forceinline__ void load_set(
    v16bf (&ah)[4], v16bf (&al)[4], v16bf (&bh)[2], v16bf (&bl)[2],
    const char* __restrict__ xh, const char* __restrict__ xl,
    const char* __restrict__ wh, const char* __restrict__ wl,
    const unsigned (&aoff)[4], const unsigned (&boff)[2], unsigned d) {
#pragma unroll
  for (int i = 0; i < 4; ++i) {
    ah[i] = ldfrag(xh, aoff[i] + d);
    al[i] = ldfrag(xl, aoff[i] + d);
  }
#pragma unroll
  for (int j = 0; j < 2; ++j) {
    bh[j] = ldfrag(wh, boff[j] + d);
    bl[j] = ldfrag(wl, boff[j] + d);
  }
}

static __device__ __forceinline__ void wmma_set(
    v8f (&acc)[4][2], const v16bf (&ah)[4], const v16bf (&al)[4],
    const v16bf (&bh)[2], const v16bf (&bl)[2]) {
#pragma unroll
  for (int i = 0; i < 4; ++i) {
#pragma unroll
    for (int j = 0; j < 2; ++j) {
      acc[i][j] = __builtin_amdgcn_wmma_f32_16x16x32_bf16(
          false, ah[i], false, bh[j], (short)0, acc[i][j], false, false);
      acc[i][j] = __builtin_amdgcn_wmma_f32_16x16x32_bf16(
          false, al[i], false, bh[j], (short)0, acc[i][j], false, false);
      acc[i][j] = __builtin_amdgcn_wmma_f32_16x16x32_bf16(
          false, ah[i], false, bl[j], (short)0, acc[i][j], false, false);
    }
  }
}

// ---------------------------------------------------------------------------
// Main GEMM: C[m,n] = sum_k x[m,k] * W'[n,k] + b[n]
// Block 256 = 8 waves (2 M x 4 N), wave tile 64x32, block tile 128x128.
// Register double-buffered over 32-wide k-chunks; 24 WMMAs per chunk.
// ---------------------------------------------------------------------------
__global__ __launch_bounds__(256) void tinylora_gemm(
    const bf16_t* __restrict__ x_hi, const bf16_t* __restrict__ x_lo,
    const bf16_t* __restrict__ w_hi, const bf16_t* __restrict__ w_lo,
    const float* __restrict__ bias, float* __restrict__ out) {
  const int lane = threadIdx.x & 31;
  const int wave = threadIdx.x >> 5;
  const int wm = wave >> 2;                 // 0..1
  const int wn = wave & 3;                  // 0..3
  const int m_w = blockIdx.y * 128 + wm * 64;
  const int n_w = blockIdx.x * 128 + wn * 32;
  const int lr = lane & 15;
  const int khalf = (lane & 16) ? 8 : 0;

  // 32-bit byte offsets (max ~64MB): enables saddr + voffset addressing.
  unsigned aoff[4], boff[2];
#pragma unroll
  for (int i = 0; i < 4; ++i)
    aoff[i] = (unsigned)((m_w + i * 16 + lr) * K_DIM + khalf) * 2u;
#pragma unroll
  for (int j = 0; j < 2; ++j)
    boff[j] = (unsigned)((n_w + j * 16 + lr) * K_DIM + khalf) * 2u;

  const char* xh = (const char*)x_hi;
  const char* xl = (const char*)x_lo;
  const char* wh = (const char*)w_hi;
  const char* wl = (const char*)w_lo;

  v8f acc[4][2];
#pragma unroll
  for (int i = 0; i < 4; ++i)
#pragma unroll
    for (int j = 0; j < 2; ++j) acc[i][j] = {};

  v16bf ah0[4], al0[4], bh0[2], bl0[2];
  v16bf ah1[4], al1[4], bh1[2], bl1[2];

  // Prologue: chunk k=0 into buffer 0.
  load_set(ah0, al0, bh0, bl0, xh, xl, wh, wl, aoff, boff, 0u);

  // 63 double-iterations cover k = 0 .. K-96; preload next before consuming.
#pragma unroll 1
  for (int k0 = 0; k0 < K_DIM - 64; k0 += 64) {
    load_set(ah1, al1, bh1, bl1, xh, xl, wh, wl, aoff, boff, 64u);
    wmma_set(acc, ah0, al0, bh0, bl0);
    load_set(ah0, al0, bh0, bl0, xh, xl, wh, wl, aoff, boff, 128u);
    wmma_set(acc, ah1, al1, bh1, bl1);
    xh += 128; xl += 128; wh += 128; wl += 128;   // scalar pointer bumps
  }
  // Tail: buffer0 holds k=K-64; load k=K-32 and consume both.
  load_set(ah1, al1, bh1, bl1, xh, xl, wh, wl, aoff, boff, 64u);
  wmma_set(acc, ah0, al0, bh0, bl0);
  wmma_set(acc, ah1, al1, bh1, bl1);

  // Epilogue: C/D layout = 8 VGPRs; VGPR r: lanes 0-15 -> M=r, 16-31 -> M=r+8.
  // Output is written once: non-temporal stores keep L2 for the bf16 operands.
#pragma unroll
  for (int i = 0; i < 4; ++i) {
    int mrow = m_w + i * 16 + ((lane >> 4) << 3);
#pragma unroll
    for (int j = 0; j < 2; ++j) {
      int ncol = n_w + j * 16 + lr;
      float bv = bias[ncol];
      float* o = out + (size_t)mrow * N_DIM + ncol;
#pragma unroll
      for (int r = 0; r < 8; ++r)
        __builtin_nontemporal_store(acc[i][j][r] + bv, o + (size_t)r * N_DIM);
    }
  }
}

// ---------------------------------------------------------------------------
// Launch
// ---------------------------------------------------------------------------
extern "C" void kernel_launch(void* const* d_in, const int* in_sizes, int n_in,
                              void* d_out, int out_size, void* d_ws, size_t ws_size,
                              hipStream_t stream) {
  const float* x  = (const float*)d_in[0];
  const float* W  = (const float*)d_in[1];
  const float* b  = (const float*)d_in[2];
  const float* U  = (const float*)d_in[3];
  const float* S  = (const float*)d_in[4];
  const float* Vh = (const float*)d_in[5];
  const float* P  = (const float*)d_in[6];
  const float* v  = (const float*)d_in[7];
  float* out = (float*)d_out;

  const size_t nx = (size_t)M_DIM * K_DIM;   // 33,554,432
  const size_t nw = (size_t)N_DIM * K_DIM;   // 16,777,216
  bf16_t* x_hi = (bf16_t*)d_ws;
  bf16_t* x_lo = x_hi + nx;
  bf16_t* w_hi = x_lo + nx;
  bf16_t* w_lo = w_hi + nw;
  float*  M2   = (float*)(w_lo + nw);        // 4096*2 floats

  tinylora_prep_m2<<<N_DIM / 256, 256, 0, stream>>>(U, S, P, v, M2);
  tinylora_fold_w<<<(nw / 4) / 256, 256, 0, stream>>>(W, Vh, M2, w_hi, w_lo);
  tinylora_cvt_x<<<(nx / 4) / 256, 256, 0, stream>>>(x, x_hi, x_lo);

  dim3 grid(N_DIM / 128, M_DIM / 128);       // 32 x 64
  tinylora_gemm<<<grid, 256, 0, stream>>>(x_hi, x_lo, w_hi, w_lo, b, out);
}
